// GINConvNet_601295421672
// MI455X (gfx1250) — matrix operand order repaired and checked by reading, not compiled, not run on hardware
//
#include <hip/hip_runtime.h>
#include <cstring>

// ---------------------------------------------------------------------------
// Types / constants
// ---------------------------------------------------------------------------
typedef _Float16 half_t;
typedef __attribute__((ext_vector_type(16))) _Float16 v16h;
typedef __attribute__((ext_vector_type(8)))  float    v8f;
typedef __attribute__((ext_vector_type(4)))  float    f32x4;
typedef __attribute__((ext_vector_type(8)))  _Float16 f16x8;

static const int kN  = 400000;   // nodes
static const int kE  = 1600000;  // edges
static const int kB  = 512;      // graphs / batch
static const int kLS = 200;      // smiles length (scan batch for GRU)
static const int kT  = 1000;     // protein length
static const int kFX = 78;       // node feature dim
static const int kDIM = 32;      // GIN hidden
static const int kGH = 100;      // GRU hidden
static const int kLH = 32;       // LSTM hidden

enum GemmFlags {
  F_RELU = 1, F_AF16 = 2, F_GATHER = 4, F_BN = 8,
  F_SPLIT = 16, F_A2 = 32, F_BIAS2 = 64
};

// B is ALWAYS pre-staged f16 in [N,K] ("transB") layout.
struct GemmP {
  const void*   A;          // f32 or f16 A matrix (row major, lda)
  const float*  A2;         // optional f32 matrix added elementwise to A
  const float*  gatherTab;  // embedding table (rows of width lda)
  const int*    gatherIdx;  // row index: idx[r] = gatherIdx[r*idxStride+idxOff]
  int idxOff, idxStride;
  const float*  Asplit;     // second A source for k >= Ksplit (f32, ldAs)
  int ldAs;
  const half_t* W;          // f16 weights, [N, Ksplit or K]
  const half_t* Wsplit;     // f16 second weight block, [N, K-Ksplit]
  int Ksplit;               // multiple of 32 when F_SPLIT
  const float*  bias;       // per-N bias (always present)
  const float*  bias2;      // second per-N bias (F_BIAS2)
  const float*  bng; const float* bnb; const float* bnm; const float* bnv;
  float* C;
  int M, N, K, lda, ldc;
};

__device__ __forceinline__ float sigm(float x) { return 1.f / (1.f + __expf(-x)); }

__device__ __forceinline__ void gatomic_add(float* p, float v) {
  __hip_atomic_fetch_add(p, v, __ATOMIC_RELAXED, __HIP_MEMORY_SCOPE_AGENT);
}
__device__ __forceinline__ void satomic_add(float* p, float v) {
  __hip_atomic_fetch_add(p, v, __ATOMIC_RELAXED, __HIP_MEMORY_SCOPE_WORKGROUP);
}

// ---------------------------------------------------------------------------
// B fragment: lane -> col (clamped), element j -> K = (j/8)*16 + hi*8 + j%8.
// Full chunk: two 16B f16 loads, no masking. Tail: clamp + select (no branch).
// ---------------------------------------------------------------------------
__device__ __forceinline__ v16h loadB16(const half_t* wb, int sw, int colC,
                                        int hi, bool full, int k0, int K,
                                        const half_t* Wfull, int strideW) {
  v16h b;
  if (full) {
    const half_t* s = wb + (size_t)colC * sw + hi * 8;
    f16x8 u0, u1;
    __builtin_memcpy(&u0, s, 16);
    __builtin_memcpy(&u1, s + 16, 16);
#pragma unroll
    for (int i = 0; i < 8; ++i) { b[i] = u0[i]; b[8 + i] = u1[i]; }
  } else {
    // K tails only occur for non-split GEMMs (split K is a multiple of 32)
#pragma unroll
    for (int j = 0; j < 16; ++j) {
      int k = k0 + ((j >> 3) << 4) + (hi << 3) + (j & 7);
      int kl = k < K ? k : K - 1;
      half_t v = Wfull[(size_t)colC * strideW + kl];
      b[j] = (k < K) ? v : (half_t)0.f;
    }
  }
  return b;
}

// ---------------------------------------------------------------------------
// Specialized WMMA GEMM: C[M,N] = post( A[M,K] @ B[K,N] + bias )
// One wave32 computes a 16x32 output (two 16x16 tiles, A fragment reused),
// K consumed in 32-chunks via v_wmma_f32_16x16x32_f16 (f32 accumulate).
// Out-of-range rows/cols compute garbage that is simply not stored; only K
// bounds are masked (select, not branch) -> branch-free inner loop.
// ---------------------------------------------------------------------------
template <int FLAGS>
__global__ __launch_bounds__(256) void wmma_gemm_k(GemmP p) {
  const bool cAF16  = (FLAGS & F_AF16)  != 0;
  const bool cGATH  = (FLAGS & F_GATHER)!= 0;
  const bool cBN    = (FLAGS & F_BN)    != 0;
  const bool cRELU  = (FLAGS & F_RELU)  != 0;
  const bool cSPLIT = (FLAGS & F_SPLIT) != 0;
  const bool cA2    = (FLAGS & F_A2)    != 0;
  const bool cB2    = (FLAGS & F_BIAS2) != 0;

  int wave = blockIdx.x * (blockDim.x >> 5) + (threadIdx.x >> 5);
  int tM = (p.M + 15) >> 4;
  int tG = (p.N + 31) >> 5;            // groups of two 16-wide tiles
  if (wave >= tM * tG) return;          // wave-uniform exit: EXEC all-1 for WMMA
  int tm = wave / tG, tg = wave - tm * tG;
  int n0 = tg * 32;
  int lane = threadIdx.x & 31;
  int hi = lane >> 4, lo = lane & 15;
  int row = tm * 16 + lo;
  int rowC = row < p.M ? row : p.M - 1;            // clamped (garbage rows unstored)
  int colC0 = (n0 + lo) < p.N ? (n0 + lo) : p.N - 1;
  int colC1 = (n0 + 16 + lo) < p.N ? (n0 + 16 + lo) : p.N - 1;
  bool hasN1 = (n0 + 16) < p.N;                    // wave-uniform

  // Resolve this lane's A row base pointers once (row pre-clamped).
  const float*  arowF = nullptr;
  const half_t* arowH = nullptr;
  const float*  arow2 = nullptr;
  const float*  arowS = nullptr;
  if (cGATH) {
    int r = p.gatherIdx[(size_t)rowC * p.idxStride + p.idxOff];
    arowF = p.gatherTab + (size_t)r * p.lda;
  } else if (cAF16) {
    arowH = (const half_t*)p.A + (size_t)rowC * p.lda;
  } else {
    arowF = (const float*)p.A + (size_t)rowC * p.lda;
    if (cA2) arow2 = p.A2 + (size_t)rowC * p.lda;
  }
  if (cSPLIT) arowS = p.Asplit + (size_t)rowC * p.ldAs;
  int strideW = cSPLIT ? p.Ksplit : p.K;
  int Ks2 = p.K - p.Ksplit;

  v8f acc0 = {0.f, 0.f, 0.f, 0.f, 0.f, 0.f, 0.f, 0.f};
  v8f acc1 = {0.f, 0.f, 0.f, 0.f, 0.f, 0.f, 0.f, 0.f};

  for (int k0 = 0; k0 < p.K; k0 += 32) {
    bool full = (k0 + 32) <= p.K;                  // wave-uniform
    bool inSplit = cSPLIT && (k0 >= p.Ksplit);     // uniform; Ksplit % 32 == 0

    // ---- A fragment: lane -> row; element j -> K = (j/8)*16 + hi*8 + j%8;
    //      two contiguous 8-element runs per lane -> b128 loads ----
    v16h a;
    if (full) {
      if (cSPLIT && inSplit) {
        const float* s = arowS + (k0 - p.Ksplit) + hi * 8;
        f32x4 x0, x1, x2, x3;
        __builtin_memcpy(&x0, s, 16);
        __builtin_memcpy(&x1, s + 4, 16);
        __builtin_memcpy(&x2, s + 16, 16);
        __builtin_memcpy(&x3, s + 20, 16);
#pragma unroll
        for (int i = 0; i < 4; ++i) {
          a[i] = (half_t)x0[i]; a[4 + i] = (half_t)x1[i];
          a[8 + i] = (half_t)x2[i]; a[12 + i] = (half_t)x3[i];
        }
      } else if (cAF16) {
        const half_t* s = arowH + k0 + hi * 8;
        f16x8 h0, h1;
        __builtin_memcpy(&h0, s, 16);
        __builtin_memcpy(&h1, s + 16, 16);
#pragma unroll
        for (int i = 0; i < 8; ++i) { a[i] = h0[i]; a[8 + i] = h1[i]; }
      } else {
        const float* s = arowF + k0 + hi * 8;
        f32x4 x0, x1, x2, x3;
        __builtin_memcpy(&x0, s, 16);
        __builtin_memcpy(&x1, s + 4, 16);
        __builtin_memcpy(&x2, s + 16, 16);
        __builtin_memcpy(&x3, s + 20, 16);
        if (cA2) {
          const float* s2 = arow2 + k0 + hi * 8;
          f32x4 y0, y1, y2, y3;
          __builtin_memcpy(&y0, s2, 16);
          __builtin_memcpy(&y1, s2 + 4, 16);
          __builtin_memcpy(&y2, s2 + 16, 16);
          __builtin_memcpy(&y3, s2 + 20, 16);
          x0 += y0; x1 += y1; x2 += y2; x3 += y3;
        }
#pragma unroll
        for (int i = 0; i < 4; ++i) {
          a[i] = (half_t)x0[i]; a[4 + i] = (half_t)x1[i];
          a[8 + i] = (half_t)x2[i]; a[12 + i] = (half_t)x3[i];
        }
      }
    } else {
      // K tail (non-split GEMMs only): clamp + select, no control flow
#pragma unroll
      for (int j = 0; j < 16; ++j) {
        int k = k0 + ((j >> 3) << 4) + (hi << 3) + (j & 7);
        int kl = k < p.K ? k : p.K - 1;
        float v;
        if (cAF16) v = (float)arowH[kl];
        else { v = arowF[kl]; if (cA2) v += arow2[kl]; }
        a[j] = (k < p.K) ? (half_t)v : (half_t)0.f;
      }
    }

    const half_t* wb = inSplit ? (p.Wsplit + (k0 - p.Ksplit)) : (p.W + k0);
    int sw = inSplit ? Ks2 : strideW;

    v16h b0 = loadB16(wb, sw, colC0, hi, full, k0, p.K, p.W, strideW);
    acc0 = __builtin_amdgcn_wmma_f32_16x16x32_f16(false, a, false, b0,
                                                  (short)0, acc0, false, false);
    if (hasN1) {
      v16h b1 = loadB16(wb, sw, colC1, hi, full, k0, p.K, p.W, strideW);
      acc1 = __builtin_amdgcn_wmma_f32_16x16x32_f16(false, a, false, b1,
                                                    (short)0, acc1, false, false);
    }
  }

  // C/D layout: VGPR i -> M = tm*16 + hi*8 + i ; N = tile_base + lo
#pragma unroll
  for (int i = 0; i < 8; ++i) {
    int m = tm * 16 + hi * 8 + i;
    if (m >= p.M) continue;
    {
      int n = n0 + lo;
      if (n < p.N) {
        float v = acc0[i] + p.bias[n];
        if (cB2) v += p.bias2[n];
        if (cRELU) v = fmaxf(v, 0.f);
        if (cBN) v = p.bng[n] * (v - p.bnm[n]) * rsqrtf(p.bnv[n] + 1e-5f) + p.bnb[n];
        p.C[(size_t)m * p.ldc + n] = v;
      }
    }
    if (hasN1) {
      int n = n0 + 16 + lo;
      if (n < p.N) {
        float v = acc1[i] + p.bias[n];
        if (cB2) v += p.bias2[n];
        if (cRELU) v = fmaxf(v, 0.f);
        if (cBN) v = p.bng[n] * (v - p.bnm[n]) * rsqrtf(p.bnv[n] + 1e-5f) + p.bnb[n];
        p.C[(size_t)m * p.ldc + n] = v;
      }
    }
  }
}

constexpr int CF_GIN1  = F_A2 | F_RELU;
constexpr int CF_GIN2  = F_RELU | F_BN;
constexpr int CF_PLAIN = F_RELU;
constexpr int CF_GRU_I = F_GATHER;
constexpr int CF_GRU_H = 0;
constexpr int CF_LSTM0 = F_GATHER | F_SPLIT | F_BIAS2;
constexpr int CF_LSTMN = F_AF16 | F_SPLIT | F_BIAS2;

// ---------------------------------------------------------------------------
// Weight staging: f32 [K,N] -> f16 [N,K]  (and plain f32->f16 copy)
// ---------------------------------------------------------------------------
__global__ void transpose_cvt(const float* __restrict__ W, half_t* __restrict__ WT,
                              int K, int N) {
  int i = blockIdx.x * blockDim.x + threadIdx.x;
  if (i >= K * N) return;
  int n = i / K, k = i - n * K;
  WT[i] = (half_t)W[(size_t)k * N + n];
}
__global__ void cvt_f16(const float* __restrict__ W, half_t* __restrict__ WT, int n) {
  int i = blockIdx.x * blockDim.x + threadIdx.x;
  if (i < n) WT[i] = (half_t)W[i];
}

// ---------------------------------------------------------------------------
// Elementwise / scatter kernels
// ---------------------------------------------------------------------------
__global__ void zero_f32(float* p, size_t n) {
  size_t i = (size_t)blockIdx.x * blockDim.x + threadIdx.x;
  if (i < n) p[i] = 0.f;
}

// agg[dst[e]][f] += h[src[e]][f]  (51MB node tensor is L2-resident: 192MB L2)
__global__ void gin_scatter(const float* __restrict__ h, const int* __restrict__ src,
                            const int* __restrict__ dst, float* __restrict__ agg,
                            long total, int F) {
  long gid = (long)blockIdx.x * blockDim.x + threadIdx.x;
  if (gid >= total) return;
  int e = (int)(gid / F), f = (int)(gid % F);
  gatomic_add(agg + (size_t)dst[e] * F + f, h[(size_t)src[e] * F + f]);
}

__global__ void pool_sum(const float* __restrict__ h, const int* __restrict__ batch,
                         float* __restrict__ pooled, int N) {
  long gid = (long)blockIdx.x * blockDim.x + threadIdx.x;
  if (gid >= (long)N * kDIM) return;
  int n = (int)(gid / kDIM), f = (int)(gid % kDIM);
  gatomic_add(pooled + (size_t)batch[n] * kDIM + f, h[(size_t)n * kDIM + f]);
}

// GRU cell (torch gate order r,z,n) + fused mean(relu(h2)) over scan-batch s
__global__ void gru_gate(const float* __restrict__ Gi, const float* __restrict__ Gh,
                         float* __restrict__ h, float* __restrict__ xc, int t, int dir) {
  int idx = blockIdx.x * blockDim.x + threadIdx.x;
  if (idx >= kLS * kGH) return;
  int s = idx / kGH, f = idx % kGH;
  float r = sigm(Gi[s * 300 + f]        + Gh[s * 300 + f]);
  float z = sigm(Gi[s * 300 + 100 + f]  + Gh[s * 300 + 100 + f]);
  float n = tanhf(Gi[s * 300 + 200 + f] + r * Gh[s * 300 + 200 + f]);
  float h2 = (1.f - z) * n + z * h[idx];
  h[idx] = h2;
  gatomic_add(xc + (size_t)t * 520 + 128 + dir * 100 + f, fmaxf(h2, 0.f) * (1.f / (float)kLS));
}

// LSTM cell (torch gate order i,f,g,o); writes h into f16 layer-output buffer
__global__ void lstm_gate(const float* __restrict__ G, float* __restrict__ c,
                          float* __restrict__ h, half_t* __restrict__ Y,
                          int time, int dir) {
  int idx = blockIdx.x * blockDim.x + threadIdx.x;
  if (idx >= kB * kLH) return;
  int b = idx / kLH, j = idx % kLH;
  float gi = G[b * 128 + j];
  float gf = G[b * 128 + 32 + j];
  float gg = G[b * 128 + 64 + j];
  float go = G[b * 128 + 96 + j];
  float c2 = sigm(gf) * c[idx] + sigm(gi) * tanhf(gg);
  float h2 = sigm(go) * tanhf(c2);
  c[idx] = c2;
  h[idx] = h2;
  Y[(size_t)time * (kB * 64) + b * 64 + dir * 32 + j] = (half_t)h2;
}

__global__ void copy_hn(const float* __restrict__ h, float* __restrict__ hn, int slot) {
  int i = blockIdx.x * blockDim.x + threadIdx.x;
  if (i < kB * kLH) hn[(size_t)slot * kB * kLH + i] = h[i];
}

// Attention per (graph b, head k<3): softmax over t, context into xc[.,328..519]
__global__ __launch_bounds__(256) void attention_k(const half_t* __restrict__ Y2,
                                                   const float* __restrict__ hn,
                                                   float* __restrict__ xc) {
  __shared__ float hid[192];
  __shared__ float ctx[192];
  __shared__ float attnS[3 * kT];
  __shared__ float rbuf[256];
  __shared__ float mx[3], sm[3];
  int b = blockIdx.x, tid = threadIdx.x;

  for (int i = tid; i < 192; i += 256) {       // faithful hn.view(-1,64,3)
    hid[i] = hn[(size_t)b * 192 + i];
    ctx[i] = 0.f;
  }
  __syncthreads();

  float lm[3] = {-1e30f, -1e30f, -1e30f};
  for (int t = tid; t < kT; t += 256) {
    const half_t* o = Y2 + (size_t)t * (kB * 64) + b * 64;
    float a0 = 0.f, a1 = 0.f, a2 = 0.f;
    for (int d = 0; d < 64; ++d) {
      float od = (float)o[d];
      a0 += od * hid[d * 3 + 0];
      a1 += od * hid[d * 3 + 1];
      a2 += od * hid[d * 3 + 2];
    }
    attnS[t * 3 + 0] = a0; attnS[t * 3 + 1] = a1; attnS[t * 3 + 2] = a2;
    lm[0] = fmaxf(lm[0], a0); lm[1] = fmaxf(lm[1], a1); lm[2] = fmaxf(lm[2], a2);
  }
  for (int k = 0; k < 3; ++k) {
    rbuf[tid] = lm[k];
    __syncthreads();
    for (int s = 128; s > 0; s >>= 1) {
      if (tid < s) rbuf[tid] = fmaxf(rbuf[tid], rbuf[tid + s]);
      __syncthreads();
    }
    if (tid == 0) mx[k] = rbuf[0];
    __syncthreads();
  }
  float ls[3] = {0.f, 0.f, 0.f};
  for (int t = tid; t < kT; t += 256) {
    ls[0] += __expf(attnS[t * 3 + 0] - mx[0]);
    ls[1] += __expf(attnS[t * 3 + 1] - mx[1]);
    ls[2] += __expf(attnS[t * 3 + 2] - mx[2]);
  }
  for (int k = 0; k < 3; ++k) {
    rbuf[tid] = ls[k];
    __syncthreads();
    for (int s = 128; s > 0; s >>= 1) {
      if (tid < s) rbuf[tid] += rbuf[tid + s];
      __syncthreads();
    }
    if (tid == 0) sm[k] = rbuf[0];
    __syncthreads();
  }
  for (int t = tid; t < kT; t += 256) {
    const half_t* o = Y2 + (size_t)t * (kB * 64) + b * 64;
    float w0 = __expf(attnS[t * 3 + 0] - mx[0]) / sm[0];
    float w1 = __expf(attnS[t * 3 + 1] - mx[1]) / sm[1];
    float w2 = __expf(attnS[t * 3 + 2] - mx[2]) / sm[2];
    for (int d = 0; d < 64; ++d) {
      float od = (float)o[d];
      satomic_add(&ctx[d * 3 + 0], od * w0);
      satomic_add(&ctx[d * 3 + 1], od * w1);
      satomic_add(&ctx[d * 3 + 2], od * w2);
    }
  }
  __syncthreads();
  for (int i = tid; i < 192; i += 256) xc[(size_t)b * 520 + 328 + i] = ctx[i];
}

__global__ void final_out(const float* __restrict__ f2, const float* __restrict__ Wout,
                          const float* __restrict__ bout, float* __restrict__ out) {
  __shared__ float r[256];
  int b = blockIdx.x, tid = threadIdx.x;
  float s = 0.f;
  for (int i = tid; i < 512; i += 256) s += f2[(size_t)b * 512 + i] * Wout[i];
  r[tid] = s;
  __syncthreads();
  for (int st = 128; st > 0; st >>= 1) {
    if (tid < st) r[tid] += r[tid + st];
    __syncthreads();
  }
  if (tid == 0) out[b] = r[0] + bout[0];
}

// ---------------------------------------------------------------------------
// Host orchestration
// ---------------------------------------------------------------------------
static inline GemmP mkP() { GemmP p; memset(&p, 0, sizeof(p)); return p; }

static inline void launch_gemm(hipStream_t s, const GemmP& p, int flags) {
  int tiles = ((p.M + 15) / 16) * ((p.N + 31) / 32);
  unsigned grid = (unsigned)((tiles + 7) / 8);   // 8 waves per 256-thread block
  switch (flags) {
    case CF_GIN1:  wmma_gemm_k<CF_GIN1><<<grid, 256, 0, s>>>(p);  break;
    case CF_GIN2:  wmma_gemm_k<CF_GIN2><<<grid, 256, 0, s>>>(p);  break;
    case CF_PLAIN: wmma_gemm_k<CF_PLAIN><<<grid, 256, 0, s>>>(p); break;
    case CF_GRU_I: wmma_gemm_k<CF_GRU_I><<<grid, 256, 0, s>>>(p); break;
    case CF_GRU_H: wmma_gemm_k<CF_GRU_H><<<grid, 256, 0, s>>>(p); break;
    case CF_LSTM0: wmma_gemm_k<CF_LSTM0><<<grid, 256, 0, s>>>(p); break;
    case CF_LSTMN: wmma_gemm_k<CF_LSTMN><<<grid, 256, 0, s>>>(p); break;
    default: break;
  }
}
static inline void launch_zero(hipStream_t s, float* p, size_t n) {
  unsigned grid = (unsigned)((n + 255) / 256);
  zero_f32<<<grid, 256, 0, s>>>(p, n);
}

extern "C" void kernel_launch(void* const* d_in, const int* in_sizes, int n_in,
                              void* d_out, int out_size, void* d_ws, size_t ws_size,
                              hipStream_t stream) {
  (void)in_sizes; (void)n_in; (void)out_size; (void)ws_size;

  // ---- input map (setup_inputs dict order, leaves flattened) ----
  const float* x      = (const float*)d_in[0];
  const int*   edge   = (const int*)d_in[1];
  const int*   srcI   = edge;
  const int*   dstI   = edge + kE;
  const int*   batch  = (const int*)d_in[2];
  const int*   smile  = (const int*)d_in[3];
  const int*   target = (const int*)d_in[4];
  auto IN = [&](int i) { return (const float*)d_in[i]; };
  const float* embed_smile = IN(77);
  const float* embed_xt    = IN(78);
  const float* Wxd = IN(79); const float* bxd = IN(80);
  const float* Wf1 = IN(81); const float* bf1 = IN(82);
  const float* Wf2 = IN(83); const float* bf2 = IN(84);
  const float* Wout = IN(85); const float* bout = IN(86);
  float* out = (float*)d_out;

  // ---- workspace carve ----
  char* wp = (char*)d_ws;
  auto carve = [&](size_t bytes) -> void* {
    void* p = (void*)wp;
    wp += (bytes + 255) & ~(size_t)255;
    return p;
  };
  float* agg    = (float*)carve((size_t)kN * kFX * 4);
  float* h0     = (float*)carve((size_t)kN * kDIM * 4);
  float* h1     = (float*)carve((size_t)kN * kDIM * 4);
  float* t1     = (float*)carve((size_t)kN * kDIM * 4);
  float* pooled = (float*)carve((size_t)kB * kDIM * 4);
  float* xc     = (float*)carve((size_t)kB * 520 * 4);
  float* f1     = (float*)carve((size_t)kB * 1024 * 4);
  float* f2     = (float*)carve((size_t)kB * 512 * 4);
  float* Gi     = (float*)carve((size_t)kLS * 300 * 4);
  float* Gh     = (float*)carve((size_t)kLS * 300 * 4);
  float* hgru   = (float*)carve((size_t)kLS * kGH * 4);
  float* G      = (float*)carve((size_t)kB * 128 * 4);
  float* cst    = (float*)carve((size_t)kB * kLH * 4);
  float* hst    = (float*)carve((size_t)kB * kLH * 4);
  float* hn     = (float*)carve((size_t)6 * kB * kLH * 4);
  half_t* Yb0   = (half_t*)carve((size_t)kT * kB * 64 * 2);
  half_t* Yb1   = (half_t*)carve((size_t)kT * kB * 64 * 2);

  // f16 weight staging buffers ([N,K] layout)
  half_t* ginW1T[5]; half_t* ginW2T[5];
  for (int l = 0; l < 5; ++l) {
    int F = l ? kDIM : kFX;
    ginW1T[l] = (half_t*)carve((size_t)kDIM * F * 2);
    ginW2T[l] = (half_t*)carve((size_t)kDIM * kDIM * 2);
  }
  half_t* WxdT = (half_t*)carve((size_t)128 * kDIM * 2);
  half_t* Wf1T = (half_t*)carve((size_t)1024 * 520 * 2);
  half_t* Wf2T = (half_t*)carve((size_t)512 * 1024 * 2);
  half_t* gruWih16[2]; half_t* gruWhh16[2];
  for (int d = 0; d < 2; ++d) {
    gruWih16[d] = (half_t*)carve((size_t)300 * kGH * 2);
    gruWhh16[d] = (half_t*)carve((size_t)300 * kGH * 2);
  }
  half_t* lstmWih16[6]; half_t* lstmWhh16[6];
  for (int i = 0; i < 6; ++i) {
    int in_sz = (i < 2) ? 128 : 64;
    lstmWih16[i] = (half_t*)carve((size_t)128 * in_sz * 2);
    lstmWhh16[i] = (half_t*)carve((size_t)128 * kLH * 2);
  }

  // ---- stage weights (once per call, in-stream) ----
  auto tcvt = [&](const float* W, half_t* WT, int K, int N) {
    int tot = K * N;
    transpose_cvt<<<(unsigned)((tot + 255) / 256), 256, 0, stream>>>(W, WT, K, N);
  };
  auto ccvt = [&](const float* W, half_t* WT, int n) {
    cvt_f16<<<(unsigned)((n + 255) / 256), 256, 0, stream>>>(W, WT, n);
  };
  for (int l = 0; l < 5; ++l) {
    int F = l ? kDIM : kFX;
    tcvt(IN(5 + 4 * l), ginW1T[l], F, kDIM);
    tcvt(IN(7 + 4 * l), ginW2T[l], kDIM, kDIM);
  }
  tcvt(Wxd, WxdT, kDIM, 128);
  tcvt(Wf1, Wf1T, 520, 1024);
  tcvt(Wf2, Wf2T, 1024, 512);
  for (int d = 0; d < 2; ++d) {
    ccvt(IN(45 + 4 * d), gruWih16[d], 300 * kGH);
    ccvt(IN(46 + 4 * d), gruWhh16[d], 300 * kGH);
  }
  for (int i = 0; i < 6; ++i) {
    int in_sz = (i < 2) ? 128 : 64;
    ccvt(IN(53 + 4 * i), lstmWih16[i], 128 * in_sz);
    ccvt(IN(54 + 4 * i), lstmWhh16[i], 128 * kLH);
  }

  launch_zero(stream, xc, (size_t)kB * 520);   // GRU branch accumulates into xc

  // ================= GIN branch (5 layers) =================
  const float* hin = x;
  float* houts[2] = {h0, h1};
  for (int l = 0; l < 5; ++l) {
    int F = l ? kDIM : kFX;
    launch_zero(stream, agg, (size_t)kN * F);
    long tot = (long)kE * F;
    gin_scatter<<<(unsigned)((tot + 255) / 256), 256, 0, stream>>>(hin, srcI, dstI, agg, tot, F);

    GemmP p1 = mkP();                 // t1 = relu((h+agg) @ W1 + b1)
    p1.A = hin; p1.A2 = agg;
    p1.W = ginW1T[l]; p1.bias = IN(6 + 4 * l);
    p1.C = t1; p1.M = kN; p1.N = kDIM; p1.K = F; p1.lda = F; p1.ldc = kDIM;
    launch_gemm(stream, p1, CF_GIN1);

    GemmP p2 = mkP();                 // h' = BN(relu(t1 @ W2 + b2))
    p2.A = t1;
    p2.W = ginW2T[l]; p2.bias = IN(8 + 4 * l);
    p2.bng = IN(25 + 4 * l); p2.bnb = IN(26 + 4 * l);
    p2.bnm = IN(27 + 4 * l); p2.bnv = IN(28 + 4 * l);
    p2.C = houts[l & 1]; p2.M = kN; p2.N = kDIM; p2.K = kDIM; p2.lda = kDIM; p2.ldc = kDIM;
    launch_gemm(stream, p2, CF_GIN2);
    hin = houts[l & 1];
  }

  // global_add_pool + xd = relu(pooled @ Wxd + bxd) -> xc[:, 0:128]
  launch_zero(stream, pooled, (size_t)kB * kDIM);
  {
    long tot = (long)kN * kDIM;
    pool_sum<<<(unsigned)((tot + 255) / 256), 256, 0, stream>>>(hin, batch, pooled, kN);
    GemmP px = mkP();
    px.A = pooled; px.W = WxdT; px.bias = bxd;
    px.C = xc; px.M = kB; px.N = 128; px.K = kDIM; px.lda = kDIM; px.ldc = 520;
    launch_gemm(stream, px, CF_PLAIN);
  }

  // ================= SMILES bidirectional GRU =================
  for (int dir = 0; dir < 2; ++dir) {
    launch_zero(stream, hgru, (size_t)kLS * kGH);
    const float* bih = IN(47 + 4 * dir);
    const float* bhh = IN(48 + 4 * dir);
    for (int stp = 0; stp < kB; ++stp) {
      int t = dir ? (kB - 1 - stp) : stp;
      GemmP pa = mkP();               // Gi = embed_smile[smile[t,:]] @ Wih^T + bih
      pa.gatherTab = embed_smile; pa.gatherIdx = smile;
      pa.idxStride = 1; pa.idxOff = t * kLS;
      pa.W = gruWih16[dir]; pa.bias = bih;
      pa.C = Gi; pa.M = kLS; pa.N = 300; pa.K = kGH; pa.lda = kGH; pa.ldc = 300;
      launch_gemm(stream, pa, CF_GRU_I);

      GemmP pb = mkP();               // Gh = h @ Whh^T + bhh
      pb.A = hgru; pb.W = gruWhh16[dir]; pb.bias = bhh;
      pb.C = Gh; pb.M = kLS; pb.N = 300; pb.K = kGH; pb.lda = kGH; pb.ldc = 300;
      launch_gemm(stream, pb, CF_GRU_H);

      gru_gate<<<(kLS * kGH + 255) / 256, 256, 0, stream>>>(Gi, Gh, hgru, xc, t, dir);
    }
  }

  // ================= Protein 3-layer bidirectional LSTM =================
  half_t* Yarr[2] = {Yb0, Yb1};
  for (int l = 0; l < 3; ++l) {
    int in_sz = l ? 64 : 128;
    half_t* Yout = Yarr[l & 1];
    const half_t* Yin = Yarr[(l + 1) & 1];
    for (int d = 0; d < 2; ++d) {
      launch_zero(stream, cst, (size_t)kB * kLH);
      launch_zero(stream, hst, (size_t)kB * kLH);
      int pi = 53 + (l * 2 + d) * 4;
      for (int stp = 0; stp < kT; ++stp) {
        int time = d ? (kT - 1 - stp) : stp;
        // One fused GEMM per step: G = [x_t | h] @ [Wih^T ; Whh^T] + bih + bhh
        GemmP p = mkP();
        p.M = kB; p.N = 128; p.K = in_sz + kLH; p.Ksplit = in_sz;
        p.Asplit = hst; p.ldAs = kLH;
        p.W = lstmWih16[l * 2 + d]; p.Wsplit = lstmWhh16[l * 2 + d];
        p.bias = IN(pi + 2); p.bias2 = IN(pi + 3);
        p.C = G; p.ldc = 128;
        if (l == 0) {
          p.gatherTab = embed_xt; p.gatherIdx = target;     // fused embedding
          p.idxStride = kT; p.idxOff = time; p.lda = 128;
          launch_gemm(stream, p, CF_LSTM0);
        } else {
          p.A = Yin + (size_t)time * kB * 64; p.lda = 64;
          launch_gemm(stream, p, CF_LSTMN);
        }
        lstm_gate<<<(kB * kLH + 255) / 256, 256, 0, stream>>>(G, cst, hst, Yout, time, d);
      }
      copy_hn<<<(kB * kLH + 255) / 256, 256, 0, stream>>>(hst, hn, l * 2 + d);
    }
  }

  // ================= Attention -> xc[:, 328:520] =================
  attention_k<<<kB, 256, 0, stream>>>(Yarr[0], hn, xc);   // layer-2 output in Yarr[0]

  // ================= Final MLP =================
  {
    GemmP q1 = mkP();                 // f1 = relu(xc @ Wf1 + bf1)
    q1.A = xc; q1.W = Wf1T; q1.bias = bf1;
    q1.C = f1; q1.M = kB; q1.N = 1024; q1.K = 520; q1.lda = 520; q1.ldc = 1024;
    launch_gemm(stream, q1, CF_PLAIN);

    GemmP q2 = mkP();                 // f2 = relu(f1 @ Wf2 + bf2)
    q2.A = f1; q2.W = Wf2T; q2.bias = bf2;
    q2.C = f2; q2.M = kB; q2.N = 512; q2.K = 1024; q2.lda = 1024; q2.ldc = 512;
    launch_gemm(stream, q2, CF_PLAIN);

    final_out<<<kB, 256, 0, stream>>>(f2, Wout, bout, out);
  }
}
